// InfluencePredictor_20426864459783
// MI455X (gfx1250) — matrix operand order repaired and checked by reading, not compiled, not run on hardware
//
#include <hip/hip_runtime.h>
#include <hip/hip_bf16.h>

typedef _Float16 v16h __attribute__((ext_vector_type(16)));
typedef _Float16 v8h  __attribute__((ext_vector_type(8)));
typedef float    v8f  __attribute__((ext_vector_type(8)));

#define N_NODES 100000
#define N_EDGES 1000000

// ---------------------------------------------------------------------------
// Repack a row-major f32 weight matrix W[K][N] into WMMA B-operand f16 layout.
// Tile (tk,tn) covers K=[tk*32,tk*32+32), N=[tn*16,tn*16+16).
// Per tile: 32 lanes x 16 halfs; lane L holds column N = tn*16 + (L%16),
// K elements e=0..7 at K = tk*32 + 8*(L/16) + e, e=8..15 at K = tk*32 + 16 + 8*(L/16) + (e-8).
// Out-of-range (K>=Kdim or N>=Ndim) -> 0 (gives free zero-padding for We1).
// ---------------------------------------------------------------------------
__global__ void repack_b_kernel(const float* __restrict__ W, int Kdim, int Ndim,
                                int Ktiles, int Ntiles, _Float16* __restrict__ out) {
    int tid = blockIdx.x * blockDim.x + threadIdx.x;
    int total = Ktiles * Ntiles * 32;
    if (tid >= total) return;
    int lane = tid & 31;
    int t    = tid >> 5;
    int tk = t / Ntiles;
    int tn = t % Ntiles;
    int kbase = tk * 32 + ((lane >> 4) << 3);
    int n     = tn * 16 + (lane & 15);
    v16h v;
#pragma unroll
    for (int e = 0; e < 16; ++e) {
        int k = kbase + e + ((e >> 3) << 3);
        float x = (k < Kdim && n < Ndim) ? W[(size_t)k * Ndim + n] : 0.0f;
        v[e] = (_Float16)x;
    }
    *(v16h*)(out + (size_t)tid * 16) = v;
}

// Load one packed B tile (32 bytes contiguous per lane).
__device__ __forceinline__ v16h load_b(const _Float16* __restrict__ packed,
                                       int tileIdx, int lane) {
    return *(const v16h*)(packed + ((size_t)tileIdx * 32 + lane) * 16);
}

// Load one A tile from a row-major f16 LDS buffer [16][stride].
// Lane L: row r=L%16, halfs 0..7 = K k0..k0+7, halfs 8..15 = K k0+16..k0+23,
// with k0 = kt*32 + 8*(L/16).
__device__ __forceinline__ v16h load_a_lds(const _Float16* base, int stride,
                                           int lane, int kt) {
    int r  = lane & 15;
    int k0 = kt * 32 + ((lane >> 4) << 3);
    v16h a;
    *(v8h*)&a       = *(const v8h*)(base + r * stride + k0);
    *((v8h*)&a + 1) = *(const v8h*)(base + r * stride + k0 + 16);
    return a;
}

#define WMMA_F16(a, b, c) \
    __builtin_amdgcn_wmma_f32_16x16x32_f16(false, (a), false, (b), (short)0, (c), false, false)

// ---------------------------------------------------------------------------
// Node encoder: emb = relu(X @ Wn1 + bn1) @ Wn2 + bn2   (f16 output table)
// One 16-node tile per wave; 8 waves / block.
// ---------------------------------------------------------------------------
__global__ __launch_bounds__(256) void node_enc_kernel(
    const float* __restrict__ nf,
    const _Float16* __restrict__ Wn1p, const float* __restrict__ bn1,
    const _Float16* __restrict__ Wn2p, const float* __restrict__ bn2,
    _Float16* __restrict__ emb) {
    __shared__ _Float16 sh[8 * 16 * 128];
    const int lane = threadIdx.x & 31;
    const int wid  = threadIdx.x >> 5;
    const int tile = blockIdx.x * 8 + wid;
    if (tile >= N_NODES / 16) return;   // wave-uniform guard
    const int m0 = tile * 16;
    _Float16* h1 = sh + wid * (16 * 128);
    const int r  = lane & 15;
    const int hh = lane >> 4;

    // ---- layer 1: K=128 (4 K-tiles), N=128 (8 N-tiles), A from global f32
    v8f acc[8] = {};
#pragma unroll
    for (int kt = 0; kt < 4; ++kt) {
        const float* rowp = nf + (size_t)(m0 + r) * 128 + kt * 32 + hh * 8;
        v16h a;
#pragma unroll
        for (int i = 0; i < 8; ++i) {
            a[i]     = (_Float16)rowp[i];
            a[8 + i] = (_Float16)rowp[16 + i];
        }
#pragma unroll
        for (int c = 0; c < 8; ++c) {
            v16h b = load_b(Wn1p, kt * 8 + c, lane);
            acc[c] = WMMA_F16(a, b, acc[c]);
        }
    }
    // bias + relu -> stage f16 in LDS [16][128]
#pragma unroll
    for (int c = 0; c < 8; ++c) {
        float bv = bn1[c * 16 + r];
#pragma unroll
        for (int v = 0; v < 8; ++v) {
            float x = acc[c][v] + bv;
            x = x > 0.0f ? x : 0.0f;
            h1[(v + hh * 8) * 128 + c * 16 + r] = (_Float16)x;
        }
    }

    // ---- layer 2: A from LDS
    v8f acc2[8] = {};
#pragma unroll
    for (int kt = 0; kt < 4; ++kt) {
        v16h a = load_a_lds(h1, 128, lane, kt);
#pragma unroll
        for (int c = 0; c < 8; ++c) {
            v16h b = load_b(Wn2p, kt * 8 + c, lane);
            acc2[c] = WMMA_F16(a, b, acc2[c]);
        }
    }
    // bias -> f16 node_emb (scattered b16 stores; node side is only ~4% of FLOPs)
#pragma unroll
    for (int c = 0; c < 8; ++c) {
        float bv = bn2[c * 16 + r];
#pragma unroll
        for (int v = 0; v < 8; ++v) {
            emb[(size_t)(m0 + v + hh * 8) * 128 + c * 16 + r] =
                (_Float16)(acc2[c][v] + bv);
        }
    }
}

// ---------------------------------------------------------------------------
// Edge pipeline: edge encoder + gather + predictor MLP + sigmoid.
// One 16-edge tile per wave; 4 waves / block (48 KB LDS).
// combined[16][384]: cols 0..127 src, 128..255 dst, 256..383 edge_emb.
// ---------------------------------------------------------------------------
__global__ __launch_bounds__(128) void edge_kernel(
    const float* __restrict__ ef, const int* __restrict__ eidx,
    const _Float16* __restrict__ emb,
    const _Float16* __restrict__ We1p, const float* __restrict__ be1,
    const _Float16* __restrict__ We2p, const float* __restrict__ be2,
    const _Float16* __restrict__ Wp1p, const float* __restrict__ bp1,
    const _Float16* __restrict__ Wp2p, const float* __restrict__ bp2,
    const float* __restrict__ Wp3, const float* __restrict__ bp3,
    float* __restrict__ out) {
    __shared__ _Float16 sh[4 * 16 * 384];
    const int lane = threadIdx.x & 31;
    const int wid  = threadIdx.x >> 5;
    const int tile = blockIdx.x * 4 + wid;
    if (tile >= N_EDGES / 16) return;   // wave-uniform guard
    const int m0 = tile * 16;
    _Float16* comb = sh + wid * (16 * 384);
    const int r  = lane & 15;
    const int hh = lane >> 4;

    // ---- edge encoder layer 1: K=16 zero-padded to 32, single K-tile
    v16h a0;
    {
        const float* rowp = ef + (size_t)(m0 + r) * 16 + hh * 8;
#pragma unroll
        for (int i = 0; i < 8; ++i) {
            a0[i]     = (_Float16)rowp[i];
            a0[8 + i] = (_Float16)0.0f;   // K >= 16 padding
        }
    }
    v8f acc[8] = {};
#pragma unroll
    for (int c = 0; c < 8; ++c) {
        v16h b = load_b(We1p, c, lane);
        acc[c] = WMMA_F16(a0, b, acc[c]);
    }
#pragma unroll
    for (int c = 0; c < 8; ++c) {
        float bv = be1[c * 16 + r];
#pragma unroll
        for (int v = 0; v < 8; ++v) {
            float x = acc[c][v] + bv;
            x = x > 0.0f ? x : 0.0f;
            comb[(v + hh * 8) * 384 + c * 16 + r] = (_Float16)x;   // stage in cols 0..127
        }
    }

    // ---- edge encoder layer 2 -> edge_emb into cols 256..383
    v8f acc2[8] = {};
#pragma unroll
    for (int kt = 0; kt < 4; ++kt) {
        v16h a = load_a_lds(comb, 384, lane, kt);
#pragma unroll
        for (int c = 0; c < 8; ++c) {
            v16h b = load_b(We2p, kt * 8 + c, lane);
            acc2[c] = WMMA_F16(a, b, acc2[c]);
        }
    }
#pragma unroll
    for (int c = 0; c < 8; ++c) {
        float bv = be2[c * 16 + r];
#pragma unroll
        for (int v = 0; v < 8; ++v) {
            comb[(v + hh * 8) * 384 + 256 + c * 16 + r] =
                (_Float16)(acc2[c][v] + bv);
        }
    }

    // ---- gather src/dst embeddings (L2-resident table): 32 lanes copy 32 rows
    {
        int m     = lane & 15;     // edge within tile
        int which = lane >> 4;     // 0 = src, 1 = dst
        int idx   = eidx[which * N_EDGES + m0 + m];
        const uint4* srcp = (const uint4*)(emb + (size_t)idx * 128);
        uint4* dstp = (uint4*)(comb + m * 384 + which * 128);
#pragma unroll
        for (int i = 0; i < 16; ++i) dstp[i] = srcp[i];   // 256 B per row
    }

    // ---- predictor layer 1: K=384 (12 K-tiles), N=128 (8 N-tiles)
    v8f p1[8] = {};
#pragma unroll
    for (int kt = 0; kt < 12; ++kt) {
        v16h a = load_a_lds(comb, 384, lane, kt);
#pragma unroll
        for (int c = 0; c < 8; ++c) {
            v16h b = load_b(Wp1p, kt * 8 + c, lane);
            p1[c] = WMMA_F16(a, b, p1[c]);
        }
    }
    // bias + relu -> restage in cols 0..127 (all reads of comb are complete)
#pragma unroll
    for (int c = 0; c < 8; ++c) {
        float bv = bp1[c * 16 + r];
#pragma unroll
        for (int v = 0; v < 8; ++v) {
            float x = p1[c][v] + bv;
            x = x > 0.0f ? x : 0.0f;
            comb[(v + hh * 8) * 384 + c * 16 + r] = (_Float16)x;
        }
    }

    // ---- predictor layer 2: K=128 (4 K-tiles), N=64 (4 N-tiles)
    v8f p2[4] = {};
#pragma unroll
    for (int kt = 0; kt < 4; ++kt) {
        v16h a = load_a_lds(comb, 384, lane, kt);
#pragma unroll
        for (int c = 0; c < 4; ++c) {
            v16h b = load_b(Wp2p, kt * 4 + c, lane);
            p2[c] = WMMA_F16(a, b, p2[c]);
        }
    }
#pragma unroll
    for (int c = 0; c < 4; ++c) {
        float bv = bp2[c * 16 + r];
#pragma unroll
        for (int v = 0; v < 8; ++v) {
            float x = p2[c][v] + bv;
            p2[c][v] = x > 0.0f ? x : 0.0f;
        }
    }

    // ---- final 64 -> 1 dot + sigmoid (VALU + half-wave shuffle reduction)
    float w0 = Wp3[0 * 16 + r];
    float w1 = Wp3[1 * 16 + r];
    float w2 = Wp3[2 * 16 + r];
    float w3 = Wp3[3 * 16 + r];
    float pr[8];
#pragma unroll
    for (int v = 0; v < 8; ++v) {
        pr[v] = p2[0][v] * w0 + p2[1][v] * w1 + p2[2][v] * w2 + p2[3][v] * w3;
    }
#pragma unroll
    for (int mask = 1; mask <= 8; mask <<= 1) {
#pragma unroll
        for (int v = 0; v < 8; ++v) pr[v] += __shfl_xor(pr[v], mask, 32);
    }
    float b3 = bp3[0];
    if (r < 8) {
        float val = pr[0];
#pragma unroll
        for (int v = 1; v < 8; ++v) val = (r == v) ? pr[v] : val;
        int m = r + hh * 8;
        out[m0 + m] = 1.0f / (1.0f + __expf(-(val + b3)));
    }
}

// ---------------------------------------------------------------------------
extern "C" void kernel_launch(void* const* d_in, const int* in_sizes, int n_in,
                              void* d_out, int out_size, void* d_ws, size_t ws_size,
                              hipStream_t stream) {
    const float* nf   = (const float*)d_in[0];
    const int*   eidx = (const int*)d_in[1];
    const float* ef   = (const float*)d_in[2];
    const float* Wn1  = (const float*)d_in[3];
    const float* bn1  = (const float*)d_in[4];
    const float* Wn2  = (const float*)d_in[5];
    const float* bn2  = (const float*)d_in[6];
    const float* We1  = (const float*)d_in[7];
    const float* be1  = (const float*)d_in[8];
    const float* We2  = (const float*)d_in[9];
    const float* be2  = (const float*)d_in[10];
    const float* Wp1  = (const float*)d_in[11];
    const float* bp1  = (const float*)d_in[12];
    const float* Wp2  = (const float*)d_in[13];
    const float* bp2  = (const float*)d_in[14];
    const float* Wp3  = (const float*)d_in[15];
    const float* bp3  = (const float*)d_in[16];

    // workspace layout (halfs)
    _Float16* ws   = (_Float16*)d_ws;
    _Float16* emb  = ws;                        // 100000*128 = 12,800,000 halfs
    _Float16* Wn1p = ws + 12800000;             // 4*8*512  = 16384
    _Float16* Wn2p = Wn1p + 16384;              // 4*8*512  = 16384
    _Float16* We1p = Wn2p + 16384;              // 1*8*512  =  4096
    _Float16* We2p = We1p + 4096;               // 4*8*512  = 16384
    _Float16* Wp1p = We2p + 16384;              // 12*8*512 = 49152
    _Float16* Wp2p = Wp1p + 49152;              // 4*4*512  =  8192

    auto rp = [&](const float* W, int K, int N, int Kt, int Nt, _Float16* o) {
        int total = Kt * Nt * 32;
        repack_b_kernel<<<(total + 127) / 128, 128, 0, stream>>>(W, K, N, Kt, Nt, o);
    };
    rp(Wn1, 128, 128, 4, 8, Wn1p);
    rp(Wn2, 128, 128, 4, 8, Wn2p);
    rp(We1,  16, 128, 1, 8, We1p);
    rp(We2, 128, 128, 4, 8, We2p);
    rp(Wp1, 384, 128, 12, 8, Wp1p);
    rp(Wp2, 128,  64, 4, 4, Wp2p);

    // node encoder: 6250 tiles, 8 waves/block
    node_enc_kernel<<<(6250 + 7) / 8, 256, 0, stream>>>(nf, Wn1p, bn1, Wn2p, bn2, emb);

    // edge pipeline: 62500 tiles, 4 waves/block
    edge_kernel<<<62500 / 4, 128, 0, stream>>>(ef, eidx, emb,
                                               We1p, be1, We2p, be2,
                                               Wp1p, bp1, Wp2p, bp2,
                                               Wp3, bp3, (float*)d_out);
}
